// FP8LinearWrapper_57440892616940
// MI455X (gfx1250) — 8-bit matrix operand order repaired and checked by reading, compiled, not run on hardware
//
#include <hip/hip_runtime.h>
#include <hip/hip_bf16.h>


typedef __attribute__((ext_vector_type(16))) int   v16i;
typedef __attribute__((ext_vector_type(8)))  float v8f;

typedef int v4i_vs __attribute__((vector_size(16)));
typedef __attribute__((address_space(1))) v4i_vs* gbl_b128_ptr;
typedef __attribute__((address_space(3))) v4i_vs* lds_b128_ptr;

#define FP8_MAX 448.0f
#define TILE_M 128
#define TILE_N 128
#define TILE_K 128

#if __has_builtin(__builtin_amdgcn_global_load_async_to_lds_b128) && \
    __has_builtin(__builtin_amdgcn_s_wait_asynccnt)
#define FP8LIN_ASYNC_LDS 1
#else
#define FP8LIN_ASYNC_LDS 0
#endif


__global__ void fp8lin_init(unsigned* amax_bits) {
    if (threadIdx.x < 2) amax_bits[threadIdx.x] = 0u;
}

__global__ __launch_bounds__(256) void fp8lin_amax(
    const float* __restrict__ x, size_t n4, unsigned* __restrict__ amax_bits) {
    __shared__ unsigned red[256];
    unsigned m = 0u;
    const float4* x4 = (const float4*)x;
    for (size_t i = (size_t)blockIdx.x * blockDim.x + threadIdx.x; i < n4;
         i += (size_t)gridDim.x * blockDim.x) {
        float4 v = x4[i];
        float a = fmaxf(fmaxf(fabsf(v.x), fabsf(v.y)),
                        fmaxf(fabsf(v.z), fabsf(v.w)));
        unsigned b = __float_as_uint(a);
        m = m > b ? m : b;
    }
    red[threadIdx.x] = m;
    __syncthreads();
    for (int s = 128; s > 0; s >>= 1) {
        if ((int)threadIdx.x < s) {
            unsigned o = red[threadIdx.x + s];
            if (o > red[threadIdx.x]) red[threadIdx.x] = o;
        }
        __syncthreads();
    }
    if (threadIdx.x == 0) atomicMax(amax_bits, red[0]);
}

__global__ void fp8lin_scale(const unsigned* __restrict__ amax_bits,
                             float* __restrict__ scales) {
    if (threadIdx.x == 0 && blockIdx.x == 0) {
        float ax = __uint_as_float(amax_bits[0]);
        float aw = __uint_as_float(amax_bits[1]);
        float sx = (ax == 0.0f) ? 1.0f : fminf(FP8_MAX / (ax * 1.1f), 10000.0f);
        float sw = (aw == 0.0f) ? 1.0f : fminf(FP8_MAX / (aw * 1.1f), 10000.0f);
        scales[0] = sx;
        scales[1] = sw;
        scales[2] = 1.0f / (sx * sw);
    }
}

__device__ __forceinline__ unsigned cvt_e4m3_sw(float f) {
    unsigned bits = __float_as_uint(f);
    unsigned sign = (bits >> 24) & 0x80u;
    int      exp  = (int)((bits >> 23) & 0xffu) - 127;
    unsigned man  = bits & 0x7fffffu;
    if (exp < -10) return sign;
    if (exp >= -6) {
        unsigned m   = man >> 20;
        unsigned rem = man & 0xfffffu;
        unsigned v   = (unsigned)((exp + 7) << 3) | m;
        if (rem > 0x80000u || (rem == 0x80000u && (m & 1u))) v++;
        if (v > 0x7eu) v = 0x7eu;
        return sign | v;
    }
    int      sh   = 20 + (-6 - exp);
    unsigned full = 0x800000u | man;
    unsigned m    = full >> sh;
    unsigned rem  = full & ((1u << sh) - 1u);
    unsigned half = 1u << (sh - 1);
    if (rem > half || (rem == half && (m & 1u))) m++;
    return sign | m;
}

__device__ __forceinline__ unsigned pack4_e4m3(float4 v, float s) {
    float a = fminf(fmaxf(v.x * s, -FP8_MAX), FP8_MAX);
    float b = fminf(fmaxf(v.y * s, -FP8_MAX), FP8_MAX);
    float c = fminf(fmaxf(v.z * s, -FP8_MAX), FP8_MAX);
    float d = fminf(fmaxf(v.w * s, -FP8_MAX), FP8_MAX);
#if __has_builtin(__builtin_amdgcn_cvt_pk_fp8_f32)
    int r = 0;
    r = __builtin_amdgcn_cvt_pk_fp8_f32(a, b, r, false);
    r = __builtin_amdgcn_cvt_pk_fp8_f32(c, d, r, true);
    return (unsigned)r;
#else
    return cvt_e4m3_sw(a) | (cvt_e4m3_sw(b) << 8) |
           (cvt_e4m3_sw(c) << 16) | (cvt_e4m3_sw(d) << 24);
#endif
}

__global__ __launch_bounds__(256) void fp8lin_quant(
    const float* __restrict__ in, unsigned char* __restrict__ outq,
    const float* __restrict__ scales, int which, size_t n16) {
    float s = scales[which];
    const float4* in4  = (const float4*)in;
    uint4*        out4 = (uint4*)outq;
    for (size_t i = (size_t)blockIdx.x * blockDim.x + threadIdx.x; i < n16;
         i += (size_t)gridDim.x * blockDim.x) {
        uint4 r;
        r.x = pack4_e4m3(in4[i * 4 + 0], s);
        r.y = pack4_e4m3(in4[i * 4 + 1], s);
        r.z = pack4_e4m3(in4[i * 4 + 2], s);
        r.w = pack4_e4m3(in4[i * 4 + 3], s);
        out4[i] = r;
    }
}

__device__ __forceinline__ void copy16_to_lds(unsigned char* lds_dst,
                                              const unsigned char* gsrc) {
#if FP8LIN_ASYNC_LDS
    __builtin_amdgcn_global_load_async_to_lds_b128(
        (gbl_b128_ptr)gsrc, (lds_b128_ptr)lds_dst, 0, 0);
#else
    *(uint4*)lds_dst = *(const uint4*)gsrc;
#endif
}

__device__ __forceinline__ void wait_lds_copies() {
#if FP8LIN_ASYNC_LDS
    __builtin_amdgcn_s_wait_asynccnt(0);
#endif
}

__global__ __launch_bounds__(256) void fp8lin_gemm(
    const unsigned char* __restrict__ Aq, const unsigned char* __restrict__ Bq,
    const float* __restrict__ bias, const float* __restrict__ scales,
    float* __restrict__ out, int M, int N, int K) {
    __shared__ unsigned char lA[2][TILE_M * TILE_K];
    __shared__ unsigned char lB[2][TILE_N * TILE_K];

    const int tid  = threadIdx.x;
    const int lane = tid & 31;
    const int wave = tid >> 5;
    const int lhalf = lane >> 4;
    const int lmod  = lane & 15;

    const int m0 = blockIdx.x * TILE_M;
    const int n0 = blockIdx.y * TILE_N;
    const int wm = (wave >> 1) * 32;
    const int wn = (wave & 1) * 64;

    const float out_scale = scales[2];

    v8f acc[2][4] = {};

    const int nk = K / TILE_K;

    #pragma unroll
    for (int it = 0; it < 4; ++it) {
        int c    = tid + it * 256;
        int row  = c >> 3;
        int koff = (c & 7) * 16;
        copy16_to_lds(&lA[0][row * TILE_K + koff],
                      Aq + (size_t)(m0 + row) * K + koff);
        copy16_to_lds(&lB[0][row * TILE_K + koff],
                      Bq + (size_t)(n0 + row) * K + koff);
    }
    wait_lds_copies();
    __syncthreads();

    for (int kt = 0; kt < nk; ++kt) {
        const int cur = kt & 1;
        const int k0  = kt * TILE_K;

        if (kt + 1 < nk) {
            const int kn = k0 + TILE_K;
            #pragma unroll
            for (int it = 0; it < 4; ++it) {
                int c    = tid + it * 256;
                int row  = c >> 3;
                int koff = (c & 7) * 16;
                copy16_to_lds(&lA[cur ^ 1][row * TILE_K + koff],
                              Aq + (size_t)(m0 + row) * K + kn + koff);
                copy16_to_lds(&lB[cur ^ 1][row * TILE_K + koff],
                              Bq + (size_t)(n0 + row) * K + kn + koff);
            }
            if (kt + 2 < nk) {
                __builtin_prefetch(Aq + (size_t)(m0 + (tid >> 1)) * K + kn +
                                       TILE_K + (tid & 1) * 64, 0, 1);
                __builtin_prefetch(Bq + (size_t)(n0 + (tid >> 1)) * K + kn +
                                       TILE_K + (tid & 1) * 64, 0, 1);
            }
        }

        v16i afrag[2];
        #pragma unroll
        for (int mt = 0; mt < 2; ++mt) {
            const unsigned char* base =
                &lA[cur][(wm + mt * 16 + lmod) * TILE_K + 8 * lhalf];
            unsigned long long* dst = (unsigned long long*)&afrag[mt];
            #pragma unroll
            for (int p = 0; p < 8; ++p)
                dst[p] = *(const unsigned long long*)(base + 16 * p);
        }

        v16i bfrag[4];
        #pragma unroll
        for (int nt = 0; nt < 4; ++nt) {
            const unsigned char* base =
                &lB[cur][(wn + nt * 16 + lmod) * TILE_K + 8 * lhalf];
            unsigned long long* dst = (unsigned long long*)&bfrag[nt];
            #pragma unroll
            for (int j = 0; j < 8; ++j)
                dst[j] = *(const unsigned long long*)(base + 16 * j);
        }

        #pragma unroll
        for (int mt = 0; mt < 2; ++mt)
            #pragma unroll
            for (int nt = 0; nt < 4; ++nt)
                acc[mt][nt] = __builtin_amdgcn_wmma_f32_16x16x128_fp8_fp8(
                    afrag[mt], bfrag[nt], (short)0, acc[mt][nt], false, false);

        wait_lds_copies();
        __syncthreads();
    }

    #pragma unroll
    for (int nt = 0; nt < 4; ++nt) {
        const int col = n0 + wn + nt * 16 + lmod;
        const float bv = bias[col];
        #pragma unroll
        for (int mt = 0; mt < 2; ++mt) {
            const int rbase = m0 + wm + mt * 16 + 8 * lhalf;
            #pragma unroll
            for (int r = 0; r < 8; ++r) {
                out[(size_t)(rbase + r) * N + col] =
                    acc[mt][nt][r] * out_scale + bv;
            }
        }
    }
}

extern "C" void kernel_launch(void* const* d_in, const int* in_sizes, int n_in,
                              void* d_out, int out_size, void* d_ws, size_t ws_size,
                              hipStream_t stream) {
    const float* x      = (const float*)d_in[0];
    const float* weight = (const float*)d_in[1];
    const float* bias   = (const float*)d_in[2];
    float*       out    = (float*)d_out;

    const int N = in_sizes[2];
    const int K = in_sizes[1] / N;
    const int M = in_sizes[0] / K;

    unsigned char* ws        = (unsigned char*)d_ws;
    unsigned*      amax_bits = (unsigned*)ws;
    float*         scales    = (float*)(ws + 8);
    unsigned char* x_fp8     = ws + 256;
    unsigned char* w_fp8     = ws + 256 + (size_t)M * K;

    fp8lin_init<<<1, 32, 0, stream>>>(amax_bits);

    {
        size_t n4x = (size_t)M * K / 4;
        size_t n4w = (size_t)N * K / 4;
        fp8lin_amax<<<2048, 256, 0, stream>>>(x, n4x, amax_bits + 0);
        fp8lin_amax<<<2048, 256, 0, stream>>>(weight, n4w, amax_bits + 1);
    }

    fp8lin_scale<<<1, 32, 0, stream>>>(amax_bits, scales);

    {
        size_t n16x = (size_t)M * K / 16;
        size_t n16w = (size_t)N * K / 16;
        fp8lin_quant<<<4096, 256, 0, stream>>>(x, x_fp8, scales, 0, n16x);
        fp8lin_quant<<<4096, 256, 0, stream>>>(weight, w_fp8, scales, 1, n16w);
    }

    {
        dim3 grid(M / TILE_M, N / TILE_N);
        fp8lin_gemm<<<grid, 256, 0, stream>>>(x_fp8, w_fp8, bias, scales, out,
                                              M, N, K);
    }
}
